// LSTM_31542239822448
// MI455X (gfx1250) — matrix operand order repaired and checked by reading, not compiled, not run on hardware
//
#include <hip/hip_runtime.h>

// ---------------- problem constants (from reference) ----------------
#define T_STEPS 512
#define BATCH   128
#define DIN     1024         // IN == H == 1024
#define HID     1024
#define NGATE   4096         // 4*HID
#define KTOT    2048         // DIN + HID (fused K)

#if __has_builtin(__builtin_amdgcn_sched_barrier)
#define SCHED_FENCE() __builtin_amdgcn_sched_barrier(0)
#else
#define SCHED_FENCE() asm volatile("" ::: "memory")
#endif

typedef __attribute__((ext_vector_type(16))) _Float16 v16h;
typedef __attribute__((ext_vector_type(8)))  float    v8f;

union F16Frag { v16h v; uint4 q[2]; };

// A-matrix 16x32 f16 fragment: lane = {hi=lane>>4, r=lane&15},
// row M = r; halves j=0..15 map to K = hi*16 + j
// -> 16 contiguous halves (two 16B groups) at k0+hi*16 and k0+hi*16+8.
__device__ __forceinline__ v16h load_a_frag(const _Float16* __restrict__ A,
                                            int row, int k0, int hi) {
  F16Frag f;
  const _Float16* p = A + (size_t)row * DIN + k0 + hi * 16;
  f.q[0] = *reinterpret_cast<const uint4*>(p);
  f.q[1] = *reinterpret_cast<const uint4*>(p + 8);
  return f.v;
}

// B-matrix 32x16 f16 fragment from an LDS tile stored [gate][col][K=32]:
// lane = N column (lr), halves j -> K = hi*16 + j, i.e. 16 contiguous halves.
__device__ __forceinline__ v16h load_lds_frag(const _Float16* p) {
  F16Frag f;
  f.q[0] = *reinterpret_cast<const uint4*>(p);
  f.q[1] = *reinterpret_cast<const uint4*>(p + 8);
  return f.v;
}

__device__ __forceinline__ float sigmoidf_(float x) {
  return 1.0f / (1.0f + __expf(-x));
}

// One LSTM timestep: gates = [x_t, h_prev] @ Wt^T + b ; cell update fused.
// Grid: 128 blocks x 128 threads (4 waves).
//   blockIdx>>1 = h-tile (0..63), blockIdx&1 = batch half, wave = m-tile in half.
// Weights double-buffered through LDS (16MB matrix read from L2 only 2x/step);
// A fragments software-pipelined in registers; all 4 gate B fragments loaded
// before a sched_barrier so the 4 WMMAs issue behind staggered dscnt waits.
__global__ __launch_bounds__(128) void lstm_step_kernel(
    const _Float16* __restrict__ Ax,     // [B, DIN]  layer input at t (f16)
    const _Float16* __restrict__ Ah,     // [B, HID]  h_{t-1} (f16)
    const _Float16* __restrict__ Wt,     // [NGATE, KTOT] fused transposed weights
    const float*    __restrict__ b_ih,   // [NGATE]
    const float*    __restrict__ b_hh,   // [NGATE]
    float*          __restrict__ c_state,// [B, HID] f32 (read/write)
    _Float16*       __restrict__ h16_out,// [B, HID] f16 (next-step A operand)
    float*          __restrict__ hf_out_a, // nullable [B, HID] f32
    float*          __restrict__ hf_out_b, // nullable [B, HID] f32
    float*          __restrict__ c_out)    // nullable [B, HID] f32
{
  __shared__ __align__(16) _Float16 ldsB[2][4][16][32];  // 8KB double buffer

  const int tid  = threadIdx.x;
  const int lane = tid & 31;
  const int wav  = tid >> 5;             // 0..3
  const int hi   = lane >> 4;
  const int lr   = lane & 15;
  const int h0   = (blockIdx.x >> 1) * 16;            // hidden-col tile base
  const int m0   = ((blockIdx.x & 1) * 4 + wav) * 16; // batch-row tile base

  // Cooperative staging map: 128 threads x 32B = 4KB per stage.
  // thread -> (gate sq, col sc) row of the strip, 16-half chunk sj.
  const int srow = tid >> 1;             // 0..63
  const int sq   = srow >> 4;
  const int sc   = srow & 15;
  const int sj   = (tid & 1) * 16;
  const _Float16* wrow = Wt + (size_t)(sq * HID + h0 + sc) * KTOT + sj;
  _Float16* lds0 = &ldsB[0][sq][sc][sj];
  _Float16* lds1 = &ldsB[1][sq][sc][sj];

  v8f acc[4] = {};   // i, f, g, o gate accumulators (16x16 f32 each)

  // Preload weight stage 0 into LDS buffer 0, and A stage 0 into registers.
  {
    uint4 r0 = *reinterpret_cast<const uint4*>(wrow);
    uint4 r1 = *reinterpret_cast<const uint4*>(wrow + 8);
    *reinterpret_cast<uint4*>(lds0)     = r0;
    *reinterpret_cast<uint4*>(lds0 + 8) = r1;
  }
  v16h a_cur = load_a_frag(Ax, m0 + lr, 0, hi);   // stage 0 is always Ax
  __syncthreads();

  // Flattened fused K loop (stages 0..31: Ax / x@W_ih, 32..63: Ah / h@W_hh).
  // Steady state k=0..62 is branch-free; the last stage is peeled below.
#pragma unroll 1
  for (int k = 0; k < 63; ++k) {
    const int kn = k + 1;

    // --- next stage's weight loads (global -> regs), prefetch stage k+2 ---
    const _Float16* p = wrow + (size_t)kn * 32;
    uint4 n0 = *reinterpret_cast<const uint4*>(p);
    uint4 n1 = *reinterpret_cast<const uint4*>(p + 8);
    __builtin_prefetch(wrow + (size_t)(kn + 1) * 32, 0, 3);

    // --- next stage's A fragment (source switch at stage 32 = addr cndmask) ---
    const _Float16* An = (kn < 32) ? Ax : Ah;
    v16h a_nxt = load_a_frag(An, m0 + lr, (kn & 31) * 32, hi);

    // --- all 4 gate B fragments from LDS (distinct regs, one clause)... ---
    const int buf = k & 1;
    v16h bf0 = load_lds_frag(&ldsB[buf][0][lr][hi * 16]);
    v16h bf1 = load_lds_frag(&ldsB[buf][1][lr][hi * 16]);
    v16h bf2 = load_lds_frag(&ldsB[buf][2][lr][hi * 16]);
    v16h bf3 = load_lds_frag(&ldsB[buf][3][lr][hi * 16]);
    SCHED_FENCE();  // ...then 4 back-to-back WMMAs behind staggered waits
    acc[0] = __builtin_amdgcn_wmma_f32_16x16x32_f16(
        false, a_cur, false, bf0, (short)0, acc[0], false, false);
    acc[1] = __builtin_amdgcn_wmma_f32_16x16x32_f16(
        false, a_cur, false, bf1, (short)0, acc[1], false, false);
    acc[2] = __builtin_amdgcn_wmma_f32_16x16x32_f16(
        false, a_cur, false, bf2, (short)0, acc[2], false, false);
    acc[3] = __builtin_amdgcn_wmma_f32_16x16x32_f16(
        false, a_cur, false, bf3, (short)0, acc[3], false, false);

    // --- park next weight stage in the other LDS buffer ---
    _Float16* d = buf ? lds0 : lds1;
    *reinterpret_cast<uint4*>(d)     = n0;
    *reinterpret_cast<uint4*>(d + 8) = n1;
    __syncthreads();

    a_cur = a_nxt;
  }

  // Peeled final stage (k = 63, buffer 1): no staging, no barrier needed after.
  {
    v16h bf0 = load_lds_frag(&ldsB[1][0][lr][hi * 16]);
    v16h bf1 = load_lds_frag(&ldsB[1][1][lr][hi * 16]);
    v16h bf2 = load_lds_frag(&ldsB[1][2][lr][hi * 16]);
    v16h bf3 = load_lds_frag(&ldsB[1][3][lr][hi * 16]);
    SCHED_FENCE();
    acc[0] = __builtin_amdgcn_wmma_f32_16x16x32_f16(
        false, a_cur, false, bf0, (short)0, acc[0], false, false);
    acc[1] = __builtin_amdgcn_wmma_f32_16x16x32_f16(
        false, a_cur, false, bf1, (short)0, acc[1], false, false);
    acc[2] = __builtin_amdgcn_wmma_f32_16x16x32_f16(
        false, a_cur, false, bf2, (short)0, acc[2], false, false);
    acc[3] = __builtin_amdgcn_wmma_f32_16x16x32_f16(
        false, a_cur, false, bf3, (short)0, acc[3], false, false);
  }

  // Per-column biases (same for all 16 rows of the tile).
  const int   col = h0 + lr;
  const float bsi = b_ih[col]           + b_hh[col];
  const float bsf = b_ih[col +   HID]   + b_hh[col +   HID];
  const float bsg = b_ih[col + 2*HID]   + b_hh[col + 2*HID];
  const float bso = b_ih[col + 3*HID]   + b_hh[col + 3*HID];

  // C/D layout: VGPR v, lanes 0-15 -> M=v; lanes 16-31 -> M=v+8; N = lr.
#pragma unroll
  for (int v = 0; v < 8; ++v) {
    const int   row = m0 + v + 8 * hi;        // batch index
    const int   idx = row * HID + col;
    const float ig  = sigmoidf_(acc[0][v] + bsi);
    const float fg  = sigmoidf_(acc[1][v] + bsf);
    const float gg  = tanhf   (acc[2][v] + bsg);
    const float og  = sigmoidf_(acc[3][v] + bso);
    const float cn  = fg * c_state[idx] + ig * gg;
    const float hn  = og * tanhf(cn);
    c_state[idx] = cn;
    h16_out[idx] = (_Float16)hn;
    if (hf_out_a) hf_out_a[idx] = hn;
    if (hf_out_b) hf_out_b[idx] = hn;
    if (c_out)    c_out[idx]    = cn;
  }
}

// ---------------- prep / init kernels ----------------

__global__ void f32_to_f16_kernel(const float* __restrict__ in,
                                  _Float16* __restrict__ out, int n) {
  int i = blockIdx.x * blockDim.x + threadIdx.x;
  if (i < n) out[i] = (_Float16)in[i];
}

// Build fused transposed weights for one layer:
// Wt[n*KTOT + k] = (k < DIN) ? w_ih[k*NGATE + n] : w_hh[(k-DIN)*NGATE + n]
__global__ void build_wt_kernel(const float* __restrict__ w_ih,
                                const float* __restrict__ w_hh,
                                _Float16* __restrict__ Wt) {
  int idx = blockIdx.x * blockDim.x + threadIdx.x;
  if (idx >= NGATE * KTOT) return;
  int n = idx >> 11;       // / KTOT
  int k = idx & (KTOT - 1);
  float v = (k < DIN) ? w_ih[(size_t)k * NGATE + n]
                      : w_hh[(size_t)(k - DIN) * NGATE + n];
  Wt[idx] = (_Float16)v;
}

__global__ void zero_f16_kernel(_Float16* p, int n) {
  int i = blockIdx.x * blockDim.x + threadIdx.x;
  if (i < n) p[i] = (_Float16)0.0f;
}

__global__ void zero_f32_kernel(float* p, int n) {
  int i = blockIdx.x * blockDim.x + threadIdx.x;
  if (i < n) p[i] = 0.0f;
}

// ---------------- host launch ----------------

extern "C" void kernel_launch(void* const* d_in, const int* in_sizes, int n_in,
                              void* d_out, int out_size, void* d_ws, size_t ws_size,
                              hipStream_t stream) {
  (void)in_sizes; (void)n_in; (void)out_size; (void)ws_size;

  const float* x    = (const float*)d_in[0];  // [T, B, IN]
  const float* w_ih = (const float*)d_in[1];  // [L, IN, 4H]
  const float* w_hh = (const float*)d_in[2];  // [L, H, 4H]
  const float* b_ih = (const float*)d_in[3];  // [L, 4H]
  const float* b_hh = (const float*)d_in[4];  // [L, 4H]

  // Workspace carve-up (all f16 counts are even -> float stays 4B-aligned).
  const size_t SEQ = (size_t)T_STEPS * BATCH * HID;  // == T*B*DIN
  _Float16* x16   = (_Float16*)d_ws;
  _Float16* hseq0 = x16   + SEQ;                 // layer-0 h sequence (f16)
  _Float16* hseq1 = hseq0 + SEQ;                 // layer-1 h sequence (f16)
  _Float16* wt0   = hseq1 + SEQ;                 // layer-0 fused weights
  _Float16* wt1   = wt0   + (size_t)NGATE * KTOT;
  _Float16* hzero = wt1   + (size_t)NGATE * KTOT;
  float*    cstate = (float*)(hzero + (size_t)BATCH * HID);

  float* out       = (float*)d_out;
  float* out_hseq  = out;                              // [T, B, H]
  float* out_hfin  = out + SEQ;                        // [L, B, H]
  float* out_cfin  = out_hfin + (size_t)2 * BATCH * HID;

  // ---- prep ----
  {
    int n = (int)SEQ;  // 67,108,864 < 2^31
    f32_to_f16_kernel<<<(n + 255) / 256, 256, 0, stream>>>(x, x16, n);
  }
  {
    int n = NGATE * KTOT;
    build_wt_kernel<<<(n + 255) / 256, 256, 0, stream>>>(w_ih, w_hh, wt0);
    build_wt_kernel<<<(n + 255) / 256, 256, 0, stream>>>(
        w_ih + (size_t)DIN * NGATE, w_hh + (size_t)HID * NGATE, wt1);
  }
  {
    int n = BATCH * HID;
    zero_f16_kernel<<<(n + 255) / 256, 256, 0, stream>>>(hzero, n);
  }

  // ---- recurrent layers ----
  for (int l = 0; l < 2; ++l) {
    {
      int n = BATCH * HID;
      zero_f32_kernel<<<(n + 255) / 256, 256, 0, stream>>>(cstate, n);
    }
    const _Float16* inseq  = l ? hseq0 : x16;
    _Float16*       outseq = l ? hseq1 : hseq0;
    const _Float16* Wt     = l ? wt1 : wt0;
    const float*    bihl   = b_ih + (size_t)l * NGATE;
    const float*    bhhl   = b_hh + (size_t)l * NGATE;

    for (int t = 0; t < T_STEPS; ++t) {
      const _Float16* Ax  = inseq + (size_t)t * BATCH * DIN;
      const _Float16* Ah  = t ? (outseq + (size_t)(t - 1) * BATCH * HID) : hzero;
      _Float16*       h16 = outseq + (size_t)t * BATCH * HID;
      float* hfa = (l == 1) ? (out_hseq + (size_t)t * BATCH * HID) : nullptr;
      float* hfb = (t == T_STEPS - 1) ? (out_hfin + (size_t)l * BATCH * HID) : nullptr;
      float* co  = (t == T_STEPS - 1) ? (out_cfin + (size_t)l * BATCH * HID) : nullptr;

      lstm_step_kernel<<<128, 128, 0, stream>>>(
          Ax, Ah, Wt, bihl, bhhl, cstate, h16, hfa, hfb, co);
    }
  }
}
